// Seq2SeqPoint_12704513262191
// MI455X (gfx1250) — matrix operand order repaired and checked by reading, not compiled, not run on hardware
//
#include <hip/hip_runtime.h>

// ---------------------------------------------------------------------------
// Types / WMMA helpers (CDNA5 gfx1250, wave32)
// ---------------------------------------------------------------------------
typedef __bf16 bf16;
typedef __attribute__((ext_vector_type(16))) __bf16 v16bf;
typedef __attribute__((ext_vector_type(8)))  float  v8f;

#define WMMA_BF16(Af, Bf, Cf) \
  __builtin_amdgcn_wmma_f32_16x16x32_bf16(false, (Af), false, (Bf), (short)0, (Cf), false, false)

union FragU { v16bf v; uint4 u[2]; };

// A-matrix 16x32 bf16 fragment (this code's k order, matched by B): lanes 0-15 hold row M=lane,
// K elems 0..15; lanes 16-31 hold K elems 16..31 (contiguous).
__device__ __forceinline__ v16bf load_frag_a(const bf16* __restrict__ p, int ld,
                                             int row0, int k0, int rmax) {
  int lane = threadIdx.x & 31;
  int r = row0 + (lane & 15);
  r = r > rmax ? rmax : r;
  int kh = (lane & 16) ? 16 : 0;
  const bf16* q = p + (size_t)r * ld + k0 + kh;
  FragU f;
  f.u[0] = *(const uint4*)(q);
  f.u[1] = *(const uint4*)(q + 8);
  return f.v;
}

// B-matrix 32x16 bf16 fragment: lane holds column n=lane&15; lanes 0-15 K=0..15,
// lanes 16-31 K=16..31 (contiguous). We load W row-major (N,K): row n == B column n.
__device__ __forceinline__ v16bf load_frag_b(const bf16* __restrict__ p, int ld,
                                             int row0, int k0, int rmax) {
  int lane = threadIdx.x & 31;
  int r = row0 + (lane & 15);
  r = r > rmax ? rmax : r;
  int kh = (lane & 16) ? 16 : 0;
  const bf16* q = p + (size_t)r * ld + k0 + kh;
  FragU f;
  f.u[0] = *(const uint4*)(q);
  f.u[1] = *(const uint4*)(q + 8);
  return f.v;
}

__device__ __forceinline__ float sigf(float x) { return 1.f / (1.f + __expf(-x)); }

// ---------------------------------------------------------------------------
// Small utility kernels
// ---------------------------------------------------------------------------
__launch_bounds__(256)
__global__ void cvt_f32_bf16(const float* __restrict__ src, bf16* __restrict__ dst, int n) {
  int i = (blockIdx.x * 256 + threadIdx.x) * 4;
  if (i + 3 < n) {
    float4 v = *(const float4*)(src + i);
    union { bf16 h[4]; uint2 u; } t;
    t.h[0] = (bf16)v.x; t.h[1] = (bf16)v.y; t.h[2] = (bf16)v.z; t.h[3] = (bf16)v.w;
    *(uint2*)(dst + i) = t.u;
  } else {
    for (; i < n; ++i) dst[i] = (bf16)src[i];
  }
}

__launch_bounds__(256)
__global__ void gather_embed(const int* __restrict__ tokens, const float* __restrict__ emb,
                             bf16* __restrict__ dst, int rows) {
  int tid = blockIdx.x * 256 + threadIdx.x;
  if (tid >= rows * 128) return;          // D/4 = 128 float4 per row
  int row = tid >> 7, d4 = (tid & 127) << 2;
  int tok = tokens[row];
  float4 v = *(const float4*)(emb + (size_t)tok * 512 + d4);
  union { bf16 h[4]; uint2 u; } t;
  t.h[0] = (bf16)v.x; t.h[1] = (bf16)v.y; t.h[2] = (bf16)v.z; t.h[3] = (bf16)v.w;
  *(uint2*)(dst + (size_t)row * 512 + d4) = t.u;
}

__launch_bounds__(256)
__global__ void build_dec_tokens(const int* __restrict__ pre_seq, const int* __restrict__ trg,
                                 int* __restrict__ toks) {
  int tid = blockIdx.x * 256 + threadIdx.x;
  if (tid >= 64 * 64) return;
  int b = tid >> 6, t = tid & 63;
  toks[tid] = (t == 0) ? pre_seq[b * 512 + 511] : trg[b * 64 + t - 1];
}

// concat(pre_h, post_h) -> bf16 (B x 2H)
__launch_bounds__(256)
__global__ void concat_h_bf16(const float* __restrict__ outs_pre, const float* __restrict__ outs_post,
                              bf16* __restrict__ cat) {
  int tid = blockIdx.x * 256 + threadIdx.x;
  if (tid >= 64 * 1024) return;
  int b = tid >> 10, j = tid & 1023;
  float v = (j < 512) ? outs_pre [((size_t)b * 512 + 511) * 512 + j]
                      : outs_post[((size_t)b * 512 + 511) * 512 + (j - 512)];
  cat[tid] = (bf16)v;
}

// score_enc[b, sc] = relu(enc_out[b,sc,:]) . w_enc   (deterministic per-thread loop)
__launch_bounds__(256)
__global__ void score_enc_kernel(const float* __restrict__ outs_pre, const float* __restrict__ outs_post,
                                 const float* __restrict__ out_w, float* __restrict__ score) {
  int tid = blockIdx.x * 256 + threadIdx.x;
  if (tid >= 64 * 1024) return;
  int b = tid >> 10, sc = tid & 1023;
  const float* h = (sc < 512) ? &outs_pre [((size_t)b * 512 + sc) * 512]
                              : &outs_post[((size_t)b * 512 + (sc - 512)) * 512];
  float s = 0.f;
  for (int j = 0; j < 512; j += 4) {
    float4 v = *(const float4*)(h + j);
    float4 w = *(const float4*)(out_w + j);
    s += fmaxf(v.x, 0.f) * w.x + fmaxf(v.y, 0.f) * w.y +
         fmaxf(v.z, 0.f) * w.z + fmaxf(v.w, 0.f) * w.w;
  }
  score[tid] = s;
}

__launch_bounds__(256)
__global__ void init_dec_h(const float* __restrict__ eh, float* __restrict__ hf0,
                           bf16* __restrict__ hb0) {
  int tid = blockIdx.x * 256 + threadIdx.x;
  if (tid >= 64 * 512) return;
  float v = eh[tid];
  hf0[tid] = v;
  hb0[tid] = (bf16)v;
}

// ---------------------------------------------------------------------------
// Big parallel GEMM: C[M,N] = act( A[M,K](bf16) @ W[N,K](bf16)^T + bias[N] )
// Block = 256 threads (8 waves), macrotile 128x128, per-wave 2x4 WMMA tiles.
// ---------------------------------------------------------------------------
__launch_bounds__(256)
__global__ void wmma_gemm_bias(const bf16* __restrict__ A, const bf16* __restrict__ W,
                               const float* __restrict__ bias, float* __restrict__ C,
                               int M, int N, int K, int act) {
  int w = threadIdx.x >> 5, lane = threadIdx.x & 31;
  int bm = blockIdx.y * 128, bn = blockIdx.x * 128;
  int mt = (w & 3) * 32;   // two 16-row tiles per wave
  int nt = (w >> 2) * 64;  // four 16-col tiles per wave
  v8f acc[2][4] = {};
  for (int k0 = 0; k0 < K; k0 += 32) {
    v16bf a0 = load_frag_a(A, K, bm + mt,      k0, M - 1);
    v16bf a1 = load_frag_a(A, K, bm + mt + 16, k0, M - 1);
#pragma unroll
    for (int j = 0; j < 4; ++j) {
      v16bf bb = load_frag_b(W, K, bn + nt + j * 16, k0, N - 1);
      acc[0][j] = WMMA_BF16(a0, bb, acc[0][j]);
      acc[1][j] = WMMA_BF16(a1, bb, acc[1][j]);
    }
  }
  int rowadd = (lane & 16) ? 8 : 0;
  int coll   = lane & 15;
#pragma unroll
  for (int i2 = 0; i2 < 2; ++i2)
#pragma unroll
    for (int j = 0; j < 4; ++j) {
      int col = bn + nt + j * 16 + coll;
      float bv = bias ? bias[col] : 0.f;
#pragma unroll
      for (int i = 0; i < 8; ++i) {
        int row = bm + mt + i2 * 16 + rowadd + i;
        if (row < M) {
          float v = acc[i2][j][i] + bv;
          if (act == 1) v = tanhf(v);
          C[(size_t)row * N + col] = v;
        }
      }
    }
}

// ---------------------------------------------------------------------------
// Persistent GRU scan (encoders). Grid: (8 slices, 2 encoders), 256 threads.
// LDS: Whh slice resident as 192 WMMA B-fragments (192KB) + gh staging (48KB).
// Grid barrier: monotonically increasing agent-scope atomic counter.
// ---------------------------------------------------------------------------
#define ENC_LDS (192 * 1024 + 48 * 1024)
__launch_bounds__(256)
__global__ void gru_scan_enc(const float* __restrict__ gi0, const float* __restrict__ gi1,
                             const bf16* __restrict__ Whh0, const bf16* __restrict__ Whh1,
                             const float* __restrict__ bhh0, const float* __restrict__ bhh1,
                             float* __restrict__ outs0, float* __restrict__ outs1,
                             bf16* __restrict__ hb, unsigned* __restrict__ ctrs, int steps) {
  extern __shared__ char smem[];
  v16bf* wfrag = (v16bf*)smem;                     // 192 frags * 32 lanes * 32B
  float* ghs   = (float*)(smem + 192 * 1024);      // 64 x 192

  int e = blockIdx.y;
  const float* gi  = e ? gi1  : gi0;
  const bf16*  Whh = e ? Whh1 : Whh0;
  const float* bhh = e ? bhh1 : bhh0;
  float*       outs = e ? outs1 : outs0;
  bf16* hbuf = hb + (size_t)e * 2 * 64 * 512;
  unsigned* ctr = ctrs + e;

  int g = blockIdx.x;            // H slice: cols [g*64, g*64+64)
  int c0 = g * 64;
  int w = threadIdx.x >> 5, lane = threadIdx.x & 31;

  // Stage Whh slice into LDS in B-fragment layout: frag f = n*16 + kt,
  // n in [0,12): gate q=n/4, 16 cols; kt in [0,16): K chunk of 32.
  for (int f = w; f < 192; f += 8) {
    int n = f >> 4, kt = f & 15;
    int q = n >> 2;
    int r = q * 512 + c0 + ((n & 3) << 4) + (lane & 15);
    int k0 = (kt << 5) + ((lane & 16) ? 16 : 0);
    FragU t;
    t.u[0] = *(const uint4*)(Whh + (size_t)r * 512 + k0);
    t.u[1] = *(const uint4*)(Whh + (size_t)r * 512 + k0 + 8);
    wfrag[f * 32 + lane] = t.v;
  }
  __syncthreads();

  int mt  = (w & 3) * 16;        // waves 0-3 / 4-7 each cover all 64 rows
  int nt0 = (w >> 2) * 6;        // 6 of 12 N-tiles per wave half

  for (int s = 0; s < steps; ++s) {
    const bf16* hr = hbuf + (size_t)(s & 1) * (64 * 512);
    bf16*       hw = hbuf + (size_t)((s + 1) & 1) * (64 * 512);

    v8f acc[6] = {};
    for (int kt = 0; kt < 16; ++kt) {
      v16bf a = load_frag_a(hr, 512, mt, kt * 32, 63);
#pragma unroll
      for (int j = 0; j < 6; ++j)
        acc[j] = WMMA_BF16(a, wfrag[((nt0 + j) * 16 + kt) * 32 + lane], acc[j]);
    }
    int rowadd = (lane & 16) ? 8 : 0;
#pragma unroll
    for (int j = 0; j < 6; ++j) {
      int colL = (nt0 + j) * 16 + (lane & 15);
#pragma unroll
      for (int i = 0; i < 8; ++i)
        ghs[(mt + rowadd + i) * 192 + colL] = acc[j][i];
    }
    __syncthreads();

    // Gate math + h update for this 64-col slice (f32)
    for (int idx = threadIdx.x; idx < 4096; idx += 256) {
      int b = idx >> 6, j = idx & 63;
      int col = c0 + j;
      float hrv = ghs[b * 192 + j]        + bhh[col];
      float hzv = ghs[b * 192 + 64 + j]   + bhh[512 + col];
      float hnv = ghs[b * 192 + 128 + j]  + bhh[1024 + col];
      const float* gp = gi + ((size_t)b * steps + s) * 1536;
      float ir = gp[col], iz = gp[512 + col], inn = gp[1024 + col];
      float hp = (s == 0) ? 0.f : outs[((size_t)b * steps + (s - 1)) * 512 + col];
      float r  = sigf(ir + hrv);
      float z  = sigf(iz + hzv);
      float nn = tanhf(inn + r * hnv);
      float hn = (1.f - z) * nn + z * hp;
      outs[((size_t)b * steps + s) * 512 + col] = hn;
      hw[b * 512 + col] = (bf16)hn;
    }
    __syncthreads();
    if (threadIdx.x == 0) {
      __threadfence();
      __hip_atomic_fetch_add(ctr, 1u, __ATOMIC_ACQ_REL, __HIP_MEMORY_SCOPE_AGENT);
      unsigned target = 8u * (unsigned)(s + 1);
      while (__hip_atomic_load(ctr, __ATOMIC_ACQUIRE, __HIP_MEMORY_SCOPE_AGENT) < target)
        __builtin_amdgcn_s_sleep(2);
    }
    __syncthreads();
  }
}

// ---------------------------------------------------------------------------
// Persistent decoder scan. Grid: 8 slices, 256 threads. Two grid-sync points
// per step (h+partials ready; logits emitted). Deterministic reductions.
// ---------------------------------------------------------------------------
#define DEC_LDS (192 * 1024 + 48 * 1024 + 16 * 1024)
__launch_bounds__(256)
__global__ void gru_scan_dec(const float* __restrict__ gi, const bf16* __restrict__ Whh,
                             const float* __restrict__ bhh, const float* __restrict__ score,
                             const float* __restrict__ w_hid, const float* __restrict__ out_b,
                             float* __restrict__ hf, bf16* __restrict__ hbuf,
                             float* __restrict__ partials, float* __restrict__ out,
                             unsigned* __restrict__ ctr) {
  extern __shared__ char smem[];
  v16bf* wfrag = (v16bf*)smem;
  float* ghs   = (float*)(smem + 192 * 1024);
  float* red   = (float*)(smem + 192 * 1024 + 48 * 1024);   // 64 x 64

  const int T = 64;
  const size_t BTS = (size_t)64 * 64 * 512;
  int g = blockIdx.x;
  int c0 = g * 64;
  int w = threadIdx.x >> 5, lane = threadIdx.x & 31;
  float ob = out_b[0];

  for (int f = w; f < 192; f += 8) {
    int n = f >> 4, kt = f & 15;
    int q = n >> 2;
    int r = q * 512 + c0 + ((n & 3) << 4) + (lane & 15);
    int k0 = (kt << 5) + ((lane & 16) ? 16 : 0);
    FragU t;
    t.u[0] = *(const uint4*)(Whh + (size_t)r * 512 + k0);
    t.u[1] = *(const uint4*)(Whh + (size_t)r * 512 + k0 + 8);
    wfrag[f * 32 + lane] = t.v;
  }
  __syncthreads();

  int mt  = (w & 3) * 16;
  int nt0 = (w >> 2) * 6;

  for (int s = 0; s < T; ++s) {
    const bf16*  hrb = hbuf + (size_t)(s & 1) * (64 * 512);
    bf16*        hwb = hbuf + (size_t)((s + 1) & 1) * (64 * 512);
    const float* hrf = hf + (size_t)(s & 1) * (64 * 512);
    float*       hwf = hf + (size_t)((s + 1) & 1) * (64 * 512);

    v8f acc[6] = {};
    for (int kt = 0; kt < 16; ++kt) {
      v16bf a = load_frag_a(hrb, 512, mt, kt * 32, 63);
#pragma unroll
      for (int j = 0; j < 6; ++j)
        acc[j] = WMMA_BF16(a, wfrag[((nt0 + j) * 16 + kt) * 32 + lane], acc[j]);
    }
    int rowadd = (lane & 16) ? 8 : 0;
#pragma unroll
    for (int j = 0; j < 6; ++j) {
      int colL = (nt0 + j) * 16 + (lane & 15);
#pragma unroll
      for (int i = 0; i < 8; ++i)
        ghs[(mt + rowadd + i) * 192 + colL] = acc[j][i];
    }
    __syncthreads();

    for (int idx = threadIdx.x; idx < 4096; idx += 256) {
      int b = idx >> 6, j = idx & 63;
      int col = c0 + j;
      float hrv = ghs[b * 192 + j]        + bhh[col];
      float hzv = ghs[b * 192 + 64 + j]   + bhh[512 + col];
      float hnv = ghs[b * 192 + 128 + j]  + bhh[1024 + col];
      const float* gp = gi + ((size_t)b * T + s) * 1536;
      float ir = gp[col], iz = gp[512 + col], inn = gp[1024 + col];
      float hp = hrf[b * 512 + col];
      float r  = sigf(ir + hrv);
      float z  = sigf(iz + hzv);
      float nn = tanhf(inn + r * hnv);
      float hn = (1.f - z) * nn + z * hp;
      hwf[b * 512 + col] = hn;
      hwb[b * 512 + col] = (bf16)hn;
      red[b * 64 + j] = fmaxf(hn, 0.f) * w_hid[col];
    }
    __syncthreads();
    if (threadIdx.x < 64) {                 // fixed-order reduction: deterministic
      int b = threadIdx.x;
      float sum = 0.f;
      for (int j = 0; j < 64; ++j) sum += red[b * 64 + j];
      partials[g * 64 + b] = sum;
    }
    __syncthreads();
    if (threadIdx.x == 0) {
      __threadfence();
      __hip_atomic_fetch_add(ctr, 1u, __ATOMIC_ACQ_REL, __HIP_MEMORY_SCOPE_AGENT);
      unsigned target = 16u * (unsigned)s + 8u;
      while (__hip_atomic_load(ctr, __ATOMIC_ACQUIRE, __HIP_MEMORY_SCOPE_AGENT) < target)
        __builtin_amdgcn_s_sleep(2);
    }
    __syncthreads();

    // logits for this WG's 128-column strip of the 2S map
    int sc0 = g * 128;
    for (int idx = threadIdx.x; idx < 64 * 128; idx += 256) {
      int b = idx >> 7, scl = idx & 127;
      int sc = sc0 + scl;
      float u = ob;
      for (int gg = 0; gg < 8; ++gg) u += partials[gg * 64 + b];
      float p = sigf(score[b * 1024 + sc] + u);
      size_t off = ((size_t)b * T + s) * 512;
      if (sc < 512) out[off + sc] = p;
      else          out[BTS + off + (sc - 512)] = p;
    }
    __syncthreads();
    if (threadIdx.x == 0) {
      __threadfence();
      __hip_atomic_fetch_add(ctr, 1u, __ATOMIC_ACQ_REL, __HIP_MEMORY_SCOPE_AGENT);
      unsigned target = 16u * (unsigned)s + 16u;
      while (__hip_atomic_load(ctr, __ATOMIC_ACQUIRE, __HIP_MEMORY_SCOPE_AGENT) < target)
        __builtin_amdgcn_s_sleep(2);
    }
    __syncthreads();
  }
}

// ---------------------------------------------------------------------------
// Host orchestration
// ---------------------------------------------------------------------------
extern "C" void kernel_launch(void* const* d_in, const int* in_sizes, int n_in,
                              void* d_out, int out_size, void* d_ws, size_t ws_size,
                              hipStream_t stream) {
  const int B = 64, S = 512, T = 64, H = 512, D = 512;
  const int G3 = 3 * H;                       // 1536
  const int BS = B * S, BT = B * T;

  const int*   pre_seq  = (const int*)  d_in[0];
  const int*   post_seq = (const int*)  d_in[1];
  const int*   trg      = (const int*)  d_in[2];
  const float* emb      = (const float*)d_in[3];
  const float* pre_Wih  = (const float*)d_in[4];
  const float* pre_Whh  = (const float*)d_in[5];
  const float* pre_bih  = (const float*)d_in[6];
  const float* pre_bhh  = (const float*)d_in[7];
  const float* post_Wih = (const float*)d_in[8];
  const float* post_Whh = (const float*)d_in[9];
  const float* post_bih = (const float*)d_in[10];
  const float* post_bhh = (const float*)d_in[11];
  const float* enc_fc_w = (const float*)d_in[12];
  const float* enc_fc_b = (const float*)d_in[13];
  const float* dec_Wih  = (const float*)d_in[14];
  const float* dec_Whh  = (const float*)d_in[15];
  const float* dec_bih  = (const float*)d_in[16];
  const float* dec_bhh  = (const float*)d_in[17];
  const float* out_w    = (const float*)d_in[18];
  const float* out_b    = (const float*)d_in[19];
  float* out = (float*)d_out;

  char* ws = (char*)d_ws;
  size_t off = 0;
  auto alloc = [&](size_t bytes) { void* p = ws + off; off = (off + bytes + 255) & ~(size_t)255; return p; };

  bf16* xpre   = (bf16*)alloc((size_t)BS * D * 2);
  bf16* xpost  = (bf16*)alloc((size_t)BS * D * 2);
  bf16* xdec   = (bf16*)alloc((size_t)BT * D * 2);
  bf16* wpih   = (bf16*)alloc((size_t)G3 * D * 2);
  bf16* wphh   = (bf16*)alloc((size_t)G3 * H * 2);
  bf16* wqih   = (bf16*)alloc((size_t)G3 * D * 2);
  bf16* wqhh   = (bf16*)alloc((size_t)G3 * H * 2);
  bf16* wdih   = (bf16*)alloc((size_t)G3 * D * 2);
  bf16* wdhh   = (bf16*)alloc((size_t)G3 * H * 2);
  bf16* wfc    = (bf16*)alloc((size_t)H * 2 * H * 2);
  float* gipre = (float*)alloc((size_t)BS * G3 * 4);
  float* gipost= (float*)alloc((size_t)BS * G3 * 4);
  float* gidec = (float*)alloc((size_t)BT * G3 * 4);
  float* opre  = (float*)alloc((size_t)B * S * H * 4);
  float* opost = (float*)alloc((size_t)B * S * H * 4);
  bf16* hb_enc = (bf16*)alloc((size_t)2 * 2 * B * H * 2);   // [enc][2][B*H]
  bf16* hb_dec = (bf16*)alloc((size_t)2 * B * H * 2);
  float* hf_dec= (float*)alloc((size_t)2 * B * H * 4);
  bf16* cath   = (bf16*)alloc((size_t)B * 2 * H * 2);
  float* ehid  = (float*)alloc((size_t)B * H * 4);
  float* score = (float*)alloc((size_t)B * 2 * S * 4);
  float* parts = (float*)alloc((size_t)8 * B * 4);
  int*   dtok  = (int*)  alloc((size_t)BT * 4);
  unsigned* ctrs = (unsigned*)alloc(256);

  // Opt into >64KB dynamic LDS for the persistent scans.
  hipFuncSetAttribute((const void*)gru_scan_enc, hipFuncAttributeMaxDynamicSharedMemorySize, ENC_LDS);
  hipFuncSetAttribute((const void*)gru_scan_dec, hipFuncAttributeMaxDynamicSharedMemorySize, DEC_LDS);

  hipMemsetAsync(ctrs, 0, 256, stream);
  hipMemsetAsync(hb_enc, 0, (size_t)2 * 2 * B * H * 2, stream);   // h0 = 0 (bf16 zero == 0x0000)

  // Weight conversions to bf16
  auto cvt = [&](const float* s, bf16* d, int n) {
    cvt_f32_bf16<<<(n / 4 + 255) / 256, 256, 0, stream>>>(s, d, n);
  };
  cvt(pre_Wih,  wpih, G3 * D);
  cvt(pre_Whh,  wphh, G3 * H);
  cvt(post_Wih, wqih, G3 * D);
  cvt(post_Whh, wqhh, G3 * H);
  cvt(dec_Wih,  wdih, G3 * D);
  cvt(dec_Whh,  wdhh, G3 * H);
  cvt(enc_fc_w, wfc,  H * 2 * H);

  // Embedding gathers (bf16 activations)
  build_dec_tokens<<<(BT + 255) / 256, 256, 0, stream>>>(pre_seq, trg, dtok);
  gather_embed<<<(BS * 128 + 255) / 256, 256, 0, stream>>>(pre_seq,  emb, xpre,  BS);
  gather_embed<<<(BS * 128 + 255) / 256, 256, 0, stream>>>(post_seq, emb, xpost, BS);
  gather_embed<<<(BT * 128 + 255) / 256, 256, 0, stream>>>(dtok,     emb, xdec,  BT);

  // Input-side GEMMs: gi = x @ Wih^T + bih   (fully parallel WMMA)
  wmma_gemm_bias<<<dim3(G3 / 128, BS / 128), 256, 0, stream>>>(xpre,  wpih, pre_bih,  gipre,  BS, G3, D, 0);
  wmma_gemm_bias<<<dim3(G3 / 128, BS / 128), 256, 0, stream>>>(xpost, wqih, post_bih, gipost, BS, G3, D, 0);
  wmma_gemm_bias<<<dim3(G3 / 128, BT / 128), 256, 0, stream>>>(xdec,  wdih, dec_bih,  gidec,  BT, G3, D, 0);

  // Both encoder scans concurrently: 8 WGs per encoder, Whh resident in LDS
  gru_scan_enc<<<dim3(8, 2), 256, ENC_LDS, stream>>>(gipre, gipost, wphh, wqhh,
                                                     pre_bhh, post_bhh, opre, opost,
                                                     hb_enc, ctrs, S);

  // score_enc and enc_hidden
  score_enc_kernel<<<(B * 1024 + 255) / 256, 256, 0, stream>>>(opre, opost, out_w, score);
  concat_h_bf16<<<(B * 1024 + 255) / 256, 256, 0, stream>>>(opre, opost, cath);
  wmma_gemm_bias<<<dim3(H / 128, 1), 256, 0, stream>>>(cath, wfc, enc_fc_b, ehid, B, H, 2 * H, 1);
  init_dec_h<<<(B * H + 255) / 256, 256, 0, stream>>>(ehid, hf_dec, hb_dec);

  // Decoder scan: emits sigmoid map straight into d_out
  gru_scan_dec<<<8, 256, DEC_LDS, stream>>>(gidec, wdhh, dec_bhh, score,
                                            out_w + H, out_b, hf_dec, hb_dec,
                                            parts, out, ctrs + 2);
}
